// GINClassifier_88742614270552
// MI455X (gfx1250) — matrix operand order repaired and checked by reading, not compiled, not run on hardware
//
#include <hip/hip_runtime.h>

typedef __attribute__((ext_vector_type(16))) _Float16 v16h;
typedef __attribute__((ext_vector_type(8)))  _Float16 v8h;
typedef __attribute__((ext_vector_type(8)))  float    v8f;

#define N_NODES 50000
#define D_IN    128
#define D_HID2  256   // 2*d_h
#define N_CLS   40
#define N_CLSP  48    // padded to 3 WMMA tiles

// ---------------------------------------------------------------------------
// Weight prep: W[K,M] f32 row-major  ->  Wt[Mpad,K] f16 (transposed, padded)
// ---------------------------------------------------------------------------
__global__ void prep_weight_kernel(const float* __restrict__ W, _Float16* __restrict__ Wt,
                                   int K, int M, int Mpad) {
    int idx = blockIdx.x * blockDim.x + threadIdx.x;
    int total = Mpad * K;
    if (idx >= total) return;
    int m = idx / K;
    int k = idx - m * K;
    Wt[(size_t)m * K + k] = (m < M) ? (_Float16)W[(size_t)k * M + m] : (_Float16)0.0f;
}

// ---------------------------------------------------------------------------
// Elementwise copy f32 (agg init: agg = x / agg2 = h1), float4 vectorized
// ---------------------------------------------------------------------------
__global__ void copy_f32x4_kernel(const float4* __restrict__ src, float4* __restrict__ dst, int n4) {
    int i = blockIdx.x * blockDim.x + threadIdx.x;
    if (i < n4) dst[i] = src[i];
}

// ---------------------------------------------------------------------------
// Edge scatter-add: one wave per edge, each lane handles 4 consecutive floats
// (float4 load = 16B/lane = 512B/wave coalesced; global_atomic_add_f32 out)
// ---------------------------------------------------------------------------
__global__ void edge_scatter_kernel(const float* __restrict__ feat, float* __restrict__ agg,
                                    const int* __restrict__ src, const int* __restrict__ dst,
                                    int n_edges) {
    int e = blockIdx.x * (blockDim.x >> 5) + (threadIdx.x >> 5);
    if (e >= n_edges) return;
    int lane = threadIdx.x & 31;
    int s = src[e];
    int d = dst[e];
    float4 v = *(const float4*)(feat + (size_t)s * D_IN + lane * 4);
    float* dp = agg + (size_t)d * D_IN + lane * 4;
    atomicAdd(dp + 0, v.x);
    atomicAdd(dp + 1, v.y);
    atomicAdd(dp + 2, v.z);
    atomicAdd(dp + 3, v.w);
}

// ---------------------------------------------------------------------------
// f32 -> f16 conversion
// ---------------------------------------------------------------------------
__global__ void cvt_f32_f16_kernel(const float* __restrict__ src, _Float16* __restrict__ dst, int n) {
    int i = blockIdx.x * blockDim.x + threadIdx.x;
    if (i < n) dst[i] = (_Float16)src[i];
}

// ---------------------------------------------------------------------------
// WMMA GEMM: out[rows,M] = act(A[rows,K] @ Wt^T + bias)
//   A  : f16 row-major [rows, K]          (K multiple of 32)
//   Wt : f16 [Mpad, K]  (transposed weight, zero-padded rows M..Mpad)
//   One wave computes one 16x16 output tile via v_wmma_f32_16x16x32_f16.
//   rows must be a multiple of 16 (50000 = 3125*16).
// ---------------------------------------------------------------------------
template <typename OutT, bool RELU>
__global__ void gemm_wmma_kernel(const _Float16* __restrict__ A,
                                 const _Float16* __restrict__ Wt,
                                 const float* __restrict__ bias,
                                 OutT* __restrict__ out,
                                 int rows, int K, int M, int Mpad) {
    int wid  = (blockIdx.x * blockDim.x + threadIdx.x) >> 5;
    int lane = threadIdx.x & 31;
    int tilesM = Mpad >> 4;
    int tileR = wid / tilesM;
    int tileC = wid - tileR * tilesM;
    if (tileR * 16 >= rows) return;          // wave-uniform exit; EXEC stays all-ones

    int hi   = (lane & 16);                  // 0 or 16
    int arow = tileR * 16 + (lane & 15);
    int col  = tileC * 16 + (lane & 15);

    // A: lanes 0-15 hold K [0..16); lanes 16-31 hold K [16..32)
    const _Float16* Ap = A  + (size_t)arow * K + hi;
    // B: lanes 0-15 hold K [0..16); lanes 16-31 hold K [16..32) for column `col`
    const _Float16* Bp = Wt + (size_t)col  * K + hi;

    v8f c = {};
    for (int k0 = 0; k0 < K; k0 += 32) {
        v8h alo = *(const v8h*)(Ap + k0);
        v8h ahi = *(const v8h*)(Ap + k0 + 8);
        v16h av;
#pragma unroll
        for (int i = 0; i < 8; ++i) { av[i] = alo[i]; av[i + 8] = ahi[i]; }
        v16h bv = *(const v16h*)(Bp + k0);
        c = __builtin_amdgcn_wmma_f32_16x16x32_f16(false, av, false, bv,
                                                   (short)0, c, false, false);
    }

    float b = (col < M) ? bias[col] : 0.0f;
    if (col < M) {
#pragma unroll
        for (int v = 0; v < 8; ++v) {
            int r = tileR * 16 + v + (hi >> 1);   // rows v (+8 for upper lanes)
            float val = c[v] + b;
            if (RELU) val = fmaxf(val, 0.0f);
            out[(size_t)r * M + col] = (OutT)val;
        }
    }
}

// ---------------------------------------------------------------------------
// Launch
// ---------------------------------------------------------------------------
extern "C" void kernel_launch(void* const* d_in, const int* in_sizes, int n_in,
                              void* d_out, int out_size, void* d_ws, size_t ws_size,
                              hipStream_t stream) {
    const float* x   = (const float*)d_in[0];
    const int*   ei  = (const int*)d_in[1];     // int32 (JAX default x64-disabled)
    const float* W11 = (const float*)d_in[2];
    const float* b11 = (const float*)d_in[3];
    const float* W12 = (const float*)d_in[4];
    const float* b12 = (const float*)d_in[5];
    const float* W21 = (const float*)d_in[6];
    const float* b21 = (const float*)d_in[7];
    const float* W22 = (const float*)d_in[8];
    const float* b22 = (const float*)d_in[9];
    const float* Wfc = (const float*)d_in[10];
    const float* bfc = (const float*)d_in[11];
    float* out = (float*)d_out;

    const int N = N_NODES, D = D_IN, H2 = D_HID2;
    const int E = in_sizes[1] / 2;
    const int* src = ei;
    const int* dst = ei + E;

    // workspace carve-out (256B aligned slices)
    char* ws = (char*)d_ws;
    size_t off = 0;
    auto carve = [&](size_t bytes) -> void* {
        void* p = ws + off;
        off = (off + bytes + 255) & ~(size_t)255;
        return p;
    };
    float*    agg  = (float*)   carve((size_t)N * D * sizeof(float));     // agg1 / agg2
    float*    h1f  = (float*)   carve((size_t)N * D * sizeof(float));     // h1 (f32 for atomics)
    _Float16* a16  = (_Float16*)carve((size_t)N * D * sizeof(_Float16)); // agg f16 / h2 f16
    _Float16* t16  = (_Float16*)carve((size_t)N * H2 * sizeof(_Float16));// hidden 256-wide f16
    _Float16* w11t = (_Float16*)carve((size_t)H2 * D * sizeof(_Float16));
    _Float16* w12t = (_Float16*)carve((size_t)D * H2 * sizeof(_Float16));
    _Float16* w21t = (_Float16*)carve((size_t)H2 * D * sizeof(_Float16));
    _Float16* w22t = (_Float16*)carve((size_t)D * H2 * sizeof(_Float16));
    _Float16* wfct = (_Float16*)carve((size_t)N_CLSP * D * sizeof(_Float16));

    const int TB = 256;
    auto gridFor = [&](long n) { return (int)((n + TB - 1) / TB); };
    auto gemmGrid = [&](int rows, int Mpad) {
        long waves = (long)(rows / 16) * (Mpad / 16);
        return (int)((waves * 32 + TB - 1) / TB);
    };

    // --- weight prep (transpose + f16) ---
    prep_weight_kernel<<<gridFor((long)H2 * D), TB, 0, stream>>>(W11, w11t, D,  H2, H2);
    prep_weight_kernel<<<gridFor((long)D * H2), TB, 0, stream>>>(W12, w12t, H2, D,  D);
    prep_weight_kernel<<<gridFor((long)H2 * D), TB, 0, stream>>>(W21, w21t, D,  H2, H2);
    prep_weight_kernel<<<gridFor((long)D * H2), TB, 0, stream>>>(W22, w22t, H2, D,  D);
    prep_weight_kernel<<<gridFor((long)N_CLSP * D), TB, 0, stream>>>(Wfc, wfct, D, N_CLS, N_CLSP);

    const long nd  = (long)N * D;
    const int edgeBlocks = (E + (TB / 32) - 1) / (TB / 32);

    // --- layer 1: aggregate(x) -> MLP -> relu ---
    copy_f32x4_kernel<<<gridFor(nd / 4), TB, 0, stream>>>((const float4*)x, (float4*)agg, (int)(nd / 4));
    edge_scatter_kernel<<<edgeBlocks, TB, 0, stream>>>(x, agg, src, dst, E);
    cvt_f32_f16_kernel<<<gridFor(nd), TB, 0, stream>>>(agg, a16, (int)nd);
    gemm_wmma_kernel<_Float16, true><<<gemmGrid(N, H2), TB, 0, stream>>>(a16, w11t, b11, t16, N, D,  H2, H2);
    gemm_wmma_kernel<float,    true><<<gemmGrid(N, D),  TB, 0, stream>>>(t16, w12t, b12, h1f, N, H2, D,  D);

    // --- layer 2: aggregate(h1) -> MLP -> relu ---
    copy_f32x4_kernel<<<gridFor(nd / 4), TB, 0, stream>>>((const float4*)h1f, (float4*)agg, (int)(nd / 4));
    edge_scatter_kernel<<<edgeBlocks, TB, 0, stream>>>(h1f, agg, src, dst, E);
    cvt_f32_f16_kernel<<<gridFor(nd), TB, 0, stream>>>(agg, a16, (int)nd);
    gemm_wmma_kernel<_Float16, true><<<gemmGrid(N, H2), TB, 0, stream>>>(a16, w21t, b21, t16, N, D,  H2, H2);
    gemm_wmma_kernel<_Float16, true><<<gemmGrid(N, D),  TB, 0, stream>>>(t16, w22t, b22, a16, N, H2, D,  D);

    // --- final classifier: h2 @ Wfc + bfc (cols padded 40 -> 48, guarded) ---
    gemm_wmma_kernel<float, false><<<gemmGrid(N, N_CLSP), TB, 0, stream>>>(a16, wfct, bfc, out, N, D, N_CLS, N_CLSP);
}
